// KimiMoEGate_68195490726075
// MI455X (gfx1250) — matrix operand order repaired and checked by reading, not compiled, not run on hardware
//
#include <hip/hip_runtime.h>
#include <hip/hip_bf16.h>
#include <math.h>

typedef __attribute__((ext_vector_type(16))) __bf16 v16bf;
typedef __attribute__((ext_vector_type(8)))  __bf16 v8bf;
typedef __attribute__((ext_vector_type(4)))  __bf16 v4bf;
typedef __attribute__((ext_vector_type(8)))  float  v8f;
typedef __attribute__((ext_vector_type(4)))  float  v4f;

#define T_ROWS  32    // rows per block (two 16-row WMMA sub-tiles)
#define KC      64    // K elements staged per barrier (2 x 32-K WMMA steps)
#define NEXP    256
#define HDIM    7168
#define NGROUPS 8
#define TOPKG   4
#define TOPK    8
#define A_STRIDE 72   // bf16 elems; 16B-aligned b128 reads, row starts on banks {0,4,..,60}
#define S_STRIDE 261  // floats; 261 % 64 == 5 -> conflict-free scatter & epilogue reads

// ---- B-fragment loaders (32x16 B, this code's contiguous K order:
//      lanes 0-15 hold K 0..15, lanes 16-31 hold K 16..31, contiguous) ----
__device__ __forceinline__ void load_b_frag(const float* __restrict__ p, v16bf& b) {
  v4f x0 = *(const v4f*)(p);
  v4f x1 = *(const v4f*)(p + 4);
  v4f x2 = *(const v4f*)(p + 8);
  v4f x3 = *(const v4f*)(p + 12);
  #pragma unroll
  for (int i = 0; i < 4; ++i) {
    b[i]      = (__bf16)x0[i];
    b[4 + i]  = (__bf16)x1[i];
    b[8 + i]  = (__bf16)x2[i];
    b[12 + i] = (__bf16)x3[i];
  }
}
__device__ __forceinline__ void load_b_frag(const __bf16* __restrict__ p, v16bf& b) {
  *((v8bf*)&b)     = *(const v8bf*)(p);
  *((v8bf*)&b + 1) = *(const v8bf*)(p + 8);
}

// ---- one-time weight fp32 -> bf16 pre-pass (runs every launch; ~11 MB traffic) ----
__global__ __launch_bounds__(256) void convert_weight_kernel(
    const float* __restrict__ w, __bf16* __restrict__ out) {
  int i = (blockIdx.x * 256 + threadIdx.x) * 4;
  v4f f = *(const v4f*)(w + i);
  v4bf o;
  #pragma unroll
  for (int j = 0; j < 4; ++j) o[j] = (__bf16)f[j];
  *(v4bf*)(out + i) = o;
}

template <typename WT>
__global__ __launch_bounds__(256) void moe_gate_kernel(
    const float* __restrict__ hidden,   // [T, H] fp32
    const WT*    __restrict__ weight,   // [E, H] fp32 or bf16
    const float* __restrict__ bias,     // [E]    fp32
    float* __restrict__ out_idx,        // [T, 8] (indices as float values)
    float* __restrict__ out_w)          // [T, 8]
{
  __shared__ __align__(16) __bf16 A_s[T_ROWS * A_STRIDE];
  __shared__ float S_s[T_ROWS * S_STRIDE];
  __shared__ float bias_s[NEXP];

  const int tid  = threadIdx.x;
  const int wave = tid >> 5;
  const int lane = tid & 31;
  const int mn   = lane & 15;   // row-in-subtile (A) / expert-in-tile (B)
  const int hv   = lane >> 4;   // half-wave select
  const int row0 = blockIdx.x * T_ROWS;

  bias_s[tid] = bias[tid];      // blockDim == NEXP == 256

  // each wave owns experts [32w, 32w+32): two 16-wide B tiles
  const int e0 = wave * 32 + mn;
  const int e1 = e0 + 16;
  const WT* wp0 = weight + (size_t)e0 * HDIM + hv * 16;
  const WT* wp1 = weight + (size_t)e1 * HDIM + hv * 16;

  // cooperative A loader: thread -> 8 consecutive floats of the 32x64 chunk
  const int arow = tid >> 3;           // 0..31
  const int acol = (tid & 7) * 8;      // 0,8,...,56
  const float* ap = hidden + (size_t)(row0 + arow) * HDIM + acol;

  v8f acc00 = {};   // rows 0-15  x e0-tile
  v8f acc01 = {};   // rows 0-15  x e1-tile
  v8f acc10 = {};   // rows 16-31 x e0-tile
  v8f acc11 = {};   // rows 16-31 x e1-tile

  for (int kb = 0; kb < HDIM; kb += KC) {
    __syncthreads();  // previous iteration's A_s reads complete
    {
      v4f f0 = *(const v4f*)(ap + kb);
      v4f f1 = *(const v4f*)(ap + kb + 4);
      #pragma unroll
      for (int i = 0; i < 4; ++i) {
        A_s[arow * A_STRIDE + acol + i]     = (__bf16)f0[i];
        A_s[arow * A_STRIDE + acol + 4 + i] = (__bf16)f1[i];
      }
    }
    __syncthreads();  // A_s chunk ready

    #pragma unroll
    for (int s = 0; s < 2; ++s) {   // two 32-K WMMA steps per staged chunk
      const int ko = s * 32;

      // A fragments, 16x32, same contiguous K order as B:
      // lane half hv: elems 0..15 = K hv*16.. (contiguous, matching B)
      v16bf a0, a1;
      *((v8bf*)&a0)     = *(const v8bf*)&A_s[mn * A_STRIDE + ko + hv * 16];
      *((v8bf*)&a0 + 1) = *(const v8bf*)&A_s[mn * A_STRIDE + ko + hv * 16 + 8];
      *((v8bf*)&a1)     = *(const v8bf*)&A_s[(16 + mn) * A_STRIDE + ko + hv * 16];
      *((v8bf*)&a1 + 1) = *(const v8bf*)&A_s[(16 + mn) * A_STRIDE + ko + hv * 16 + 8];

      v16bf b0, b1;
      load_b_frag(wp0 + kb + ko, b0);
      load_b_frag(wp1 + kb + ko, b1);

      acc00 = __builtin_amdgcn_wmma_f32_16x16x32_bf16(
          false, a0, false, b0, (short)0, acc00, false, false);
      acc01 = __builtin_amdgcn_wmma_f32_16x16x32_bf16(
          false, a0, false, b1, (short)0, acc01, false, false);
      acc10 = __builtin_amdgcn_wmma_f32_16x16x32_bf16(
          false, a1, false, b0, (short)0, acc10, false, false);
      acc11 = __builtin_amdgcn_wmma_f32_16x16x32_bf16(
          false, a1, false, b1, (short)0, acc11, false, false);
    }

    if (kb + KC < HDIM) {
      __builtin_prefetch(wp0 + kb + KC, 0, 1);  // global_prefetch_b8
      __builtin_prefetch(wp1 + kb + KC, 0, 1);
    }
  }

  // scatter accumulators to LDS logits tile:
  // elem j of lane -> local row (j + 8*hv [+16]), col = expert
  #pragma unroll
  for (int j = 0; j < 8; ++j) {
    S_s[(j + 8 * hv) * S_STRIDE + e0]      = acc00[j];
    S_s[(j + 8 * hv) * S_STRIDE + e1]      = acc01[j];
    S_s[(16 + j + 8 * hv) * S_STRIDE + e0] = acc10[j];
    S_s[(16 + j + 8 * hv) * S_STRIDE + e1] = acc11[j];
  }
  __syncthreads();

  // ---- fused gating epilogue: one thread per row (wave 0) ----
  if (tid < T_ROWS) {
    float* rowp = &S_s[tid * S_STRIDE];

    float g1[NGROUPS], g2[NGROUPS];
    #pragma unroll
    for (int g = 0; g < NGROUPS; ++g) { g1[g] = -__builtin_inff(); g2[g] = -__builtin_inff(); }

    // sigmoid (stored back, uncorrected) + per-group top-2 of corrected scores
    for (int e = 0; e < NEXP; ++e) {
      float s = 1.0f / (1.0f + expf(-rowp[e]));
      rowp[e] = s;
      float c = s + bias_s[e];
      int g = e >> 5;
      if (c > g1[g]) { g2[g] = g1[g]; g1[g] = c; }
      else if (c > g2[g]) { g2[g] = c; }
    }

    // top-4 groups (strict > == lowest-index tie-break, matching jax top_k)
    int gmask = 0;
    for (int it = 0; it < TOPKG; ++it) {
      float best = -__builtin_inff(); int bi = 0;
      for (int g = 0; g < NGROUPS; ++g) {
        if ((gmask >> g) & 1) continue;
        float v = g1[g] + g2[g];
        if (v > best) { best = v; bi = g; }
      }
      gmask |= 1 << bi;
    }

    // top-8 experts over (allowed ? corrected : 0.0), reference edge-case included
    int   ci[TOPK];
    float cw[TOPK];
    float wsum = 0.0f;
    for (int it = 0; it < TOPK; ++it) {
      float best = -__builtin_inff(); int bi = 0;
      for (int e = 0; e < NEXP; ++e) {
        bool taken = false;
        for (int k = 0; k < it; ++k) taken |= (ci[k] == e);
        if (taken) continue;
        float v = ((gmask >> (e >> 5)) & 1) ? (rowp[e] + bias_s[e]) : 0.0f;
        if (v > best) { best = v; bi = e; }
      }
      ci[it] = bi;
      float sv = rowp[bi];   // UNcorrected sigmoid score as weight
      cw[it] = sv;
      wsum += sv;
    }

    const int R = row0 + tid;
    const float inv = 2.5f / (wsum + 1e-20f);
    #pragma unroll
    for (int k = 0; k < TOPK; ++k) {
      out_idx[(size_t)R * TOPK + k] = (float)ci[k];
      out_w  [(size_t)R * TOPK + k] = cw[k] * inv;
    }
  }
}

extern "C" void kernel_launch(void* const* d_in, const int* in_sizes, int n_in,
                              void* d_out, int out_size, void* d_ws, size_t ws_size,
                              hipStream_t stream) {
  const float* hidden = (const float*)d_in[0];
  const float* weight = (const float*)d_in[1];
  const float* bias   = (const float*)d_in[2];
  const int T = in_sizes[0] / HDIM;          // 8192

  float* out   = (float*)d_out;
  float* o_idx = out;                         // first T*8: indices (as float)
  float* o_w   = out + (size_t)T * TOPK;      // next  T*8: weights

  dim3 grid(T / T_ROWS), block(256);

  const size_t wbytes = (size_t)NEXP * HDIM * sizeof(__bf16);   // 3.67 MB
  if (ws_size >= wbytes) {
    // pre-convert weight to bf16 once per launch, then bf16-B main kernel
    __bf16* wbf = (__bf16*)d_ws;
    convert_weight_kernel<<<dim3((NEXP * HDIM) / (256 * 4)), dim3(256), 0, stream>>>(
        weight, wbf);
    moe_gate_kernel<__bf16><<<grid, block, 0, stream>>>(hidden, wbf, bias, o_idx, o_w);
  } else {
    // fallback: convert B on the fly inside the GEMM
    moe_gate_kernel<float><<<grid, block, 0, stream>>>(hidden, weight, bias, o_idx, o_w);
  }
}
